// GMHA_11098195492883
// MI455X (gfx1250) — matrix operand order repaired and checked by reading, not compiled, not run on hardware
//
#include <hip/hip_runtime.h>
#include <hip/hip_bf16.h>

// ---------------------------------------------------------------------------
// Transformer-XL attention + GRU for MI455X (gfx1250, wave32, WMMA).
// All matmuls use v_wmma_f32_16x16x32_f16 (f16 in, f32 accumulate).
// LDS tiles are fed with CDNA5 async global->LDS DMA
// (global_load_async_to_lds_b128, ASYNCcnt) so WMMA overlaps the copies.
// Fragment layouts used here (k contiguous in A and B, not the ISA table order):
//   A (16x32 f16): lane m = lane&15; elem e -> K = e + (lane>=16?16:0)
//   B (32x16 f16): lane n = lane&15; elem e -> K = e + (lane>=16?16:0)
//   C (16x16 f32): lane n = lane&15; vgpr g -> M = g + (lane>=16?8:0)
// ---------------------------------------------------------------------------

typedef _Float16 h16;
typedef __attribute__((ext_vector_type(16))) _Float16 v16h;
typedef __attribute__((ext_vector_type(8)))  _Float16 v8h;
typedef __attribute__((ext_vector_type(8)))  float    v8f;

#define D_MODEL 1024
#define N_HEADS 16
#define D_HEAD  64
#define QLEN    512
#define MLEN    512
#define KLEN    1024
#define BATCH   8

__device__ __forceinline__ int a_kmap(int e, int hi) {
  // A-fragment K index for element e, lane-group hi (contiguous: K = e + 16*hi, matches B)
  return e + hi * 16;
}

__device__ __forceinline__ v8f wmma_f16(v16h a, v16h b, v8f c) {
  return __builtin_amdgcn_wmma_f32_16x16x32_f16(false, a, false, b, (short)0, c,
                                                false, false);
}

// CDNA5 async global->LDS copy, 16 bytes per lane (tracked by ASYNCcnt).
// VDST = per-lane LDS byte offset (low 32 bits of the shared pointer),
// VADDR = per-lane 64-bit global address, GV mode (saddr = off).
__device__ __forceinline__ void lds_async_b128(void* lds, const void* g) {
  asm volatile("global_load_async_to_lds_b128 %0, %1, off"
               :: "v"((unsigned int)(unsigned long long)(uintptr_t)lds),
                  "v"((unsigned long long)(uintptr_t)g)
               : "memory");
}
__device__ __forceinline__ void wait_async() {
  asm volatile("s_wait_asynccnt 0x0" ::: "memory");
}

// ---------------------------------------------------------------------------
// Prep kernels
// ---------------------------------------------------------------------------
__global__ void cvt_f32_f16(const float* __restrict__ in, h16* __restrict__ out,
                            int n) {
  int i = blockIdx.x * blockDim.x + threadIdx.x;
  if (i < n) out[i] = (h16)in[i];
}

// in: R x C f32 (row-major) -> out: C x R f16 (row-major transpose)
__global__ void cvt_transpose_f16(const float* __restrict__ in,
                                  h16* __restrict__ out, int R, int C) {
  int i = blockIdx.x * blockDim.x + threadIdx.x;
  if (i >= R * C) return;
  int r = i / C, c = i % C;
  out[(size_t)c * R + r] = (h16)in[(size_t)r * C + c];
}

// cat = [mem ; inputs] -> f16, rows = pos*BATCH+b (8192 rows x 1024 cols)
__global__ void build_cat(const float* __restrict__ memp,
                          const float* __restrict__ inp,
                          h16* __restrict__ cat, int n) {
  int i = blockIdx.x * blockDim.x + threadIdx.x;
  if (i >= n) return;
  const int half = MLEN * BATCH * D_MODEL;
  float v = (i < half) ? memp[i] : inp[i - half];
  cat[i] = (h16)v;
}

__global__ void init_h(const float* __restrict__ h0, float* __restrict__ h,
                       h16* __restrict__ hh, int n) {
  int i = blockIdx.x * blockDim.x + threadIdx.x;
  if (i < n) { h[i] = h0[i]; hh[i] = (h16)h0[i]; }
}

// ---------------------------------------------------------------------------
// Generic WMMA GEMM: C(MxN) = A(MxK,f16) * B(KxN,f16) (+bias), M,N mult of 64,
// K mult of 32. Block = 256 threads (8 waves), tile 64x64, BK=32.
// Double-buffered LDS tiles filled by async-to-LDS DMA: issue next tile's
// copies right after the barrier, run WMMA on the current tile meanwhile.
// Wave w: mi = w&3 (M subtile), nj = w>>2 with two N subtiles -> 2 WMMA/iter.
// ---------------------------------------------------------------------------
#define A_STRIDE 40   // halves; 80B rows keep b128 16B-aligned, conflict-free frag reads
#define B_STRIDE 72   // halves; 144B rows, 16B-aligned

template <bool OUT_F16>
__global__ __launch_bounds__(256) void wmma_gemm(
    const h16* __restrict__ A, const h16* __restrict__ B,
    void* __restrict__ Cout, const float* __restrict__ bias,
    int M, int N, int K) {
  __shared__ h16 a_sh[2][64 * A_STRIDE];
  __shared__ h16 b_sh[2][32 * B_STRIDE];
  const int n0 = blockIdx.x * 64, m0 = blockIdx.y * 64;
  const int tid = threadIdx.x;
  const int w = tid >> 5, lane = tid & 31;
  const int mi = w & 3, nj = w >> 2;
  const int hi = lane >> 4, ln = lane & 15;

  const int ra = tid >> 2, ca = (tid & 3) * 8;   // A staging: 64 rows x 32
  const int rb = tid >> 3, cb = (tid & 7) * 8;   // B staging: 32 rows x 64

  auto stage = [&](int buf, int k0) {
    lds_async_b128(&a_sh[buf][ra * A_STRIDE + ca],
                   &A[(size_t)(m0 + ra) * K + k0 + ca]);
    lds_async_b128(&b_sh[buf][rb * B_STRIDE + cb],
                   &B[(size_t)(k0 + rb) * N + n0 + cb]);
  };

  stage(0, 0);
  v8f acc0 = {}, acc1 = {};
  for (int k0 = 0; k0 < K; k0 += 32) {
    const int buf = (k0 >> 5) & 1;
    wait_async();        // my async copies into buf have landed
    __syncthreads();     // everyone's have; prior compute on buf^1 is done
    if (k0 + 32 < K) stage(buf ^ 1, k0 + 32);  // overlap DMA with WMMA
    v16h af, bf0, bf1;
#pragma unroll
    for (int e = 0; e < 16; ++e) {
      af[e]  = a_sh[buf][(mi * 16 + ln) * A_STRIDE + a_kmap(e, hi)];
      int kb = (hi * 16 + e) * B_STRIDE;
      bf0[e] = b_sh[buf][kb + (nj * 2 + 0) * 16 + ln];
      bf1[e] = b_sh[buf][kb + (nj * 2 + 1) * 16 + ln];
    }
    acc0 = wmma_f16(af, bf0, acc0);
    acc1 = wmma_f16(af, bf1, acc1);
  }
#pragma unroll
  for (int g = 0; g < 8; ++g) {
    int m  = m0 + mi * 16 + hi * 8 + g;
    int c0 = n0 + (nj * 2 + 0) * 16 + ln;
    int c1 = n0 + (nj * 2 + 1) * 16 + ln;
    float v0 = acc0[g] + (bias ? bias[c0] : 0.f);
    float v1 = acc1[g] + (bias ? bias[c1] : 0.f);
    if (OUT_F16) {
      ((h16*)Cout)[(size_t)m * N + c0] = (h16)v0;
      ((h16*)Cout)[(size_t)m * N + c1] = (h16)v1;
    } else {
      ((float*)Cout)[(size_t)m * N + c0] = v0;
      ((float*)Cout)[(size_t)m * N + c1] = v1;
    }
  }
}

// ---------------------------------------------------------------------------
// Fused rel-attention: per (q-tile of 32, batch, head) block.
//   scores S[i,j] = ((q_i+u)·k_j + (q_i+v)·r_{j+511-i}) / 8 , masked j>i+512
//   full-row softmax (32x1024 f32 scores live in WGP LDS), then O = P·V.
// BD rel_shift is folded in: shifted[i,j] = BDraw[i, j+QLEN-1-i]; we compute a
// 16x32 BD block with WMMA over r-rows [p0, p0+31], p0 = j0 - ibase + 496, and
// gather the diagonal S(m,n) += BD[m][n-m+15] through per-wave LDS.
// K/R/V tiles are staged with async-to-LDS DMA (per-wave tiles only need
// s_wait_asynccnt, no workgroup barrier).
// ---------------------------------------------------------------------------
#define QT  32
#define SLD 1032  // padded f32 score row stride (bank-conflict-free PV reads)

static constexpr size_t ATTN_SMEM =
    (size_t)QT * SLD * 4            // S scores
    + 2 * QT * 64 * 2               // qu, qv
    + 8 * 16 * 64 * 2               // per-wave K tiles
    + 8 * 32 * 64 * 2               // per-wave R tiles
    + 8 * 16 * 32 * 4               // per-wave BD scratch
    + 32 * 64 * 2                   // shared V tile
    + (256 + 32 + 32) * 4;          // reductions

__global__ __launch_bounds__(256) void attn_fused(
    const h16* __restrict__ qkv,   // 8192 x 3072, [q|k|v] thirds
    const h16* __restrict__ rk,    // 1024 x 1024 r_head_k
    const float* __restrict__ u, const float* __restrict__ v,
    h16* __restrict__ out) {       // 4096 x 1024 attn_vec (f16)
  extern __shared__ char smem_raw[];
  float* S    = (float*)smem_raw;
  h16*   qu   = (h16*)(S + QT * SLD);
  h16*   qv   = qu + QT * 64;
  h16*   kt   = qv + QT * 64;
  h16*   rt   = kt + 8 * 16 * 64;
  float* bd   = (float*)(rt + 8 * 32 * 64);
  h16*   vt   = (h16*)(bd + 8 * 16 * 32);
  float* red  = (float*)(vt + 32 * 64);
  float* rowm = red + 256;
  float* rows = rowm + 32;

  const int qt = blockIdx.x, b = blockIdx.y, h = blockIdx.z;
  const int i0 = qt * QT;
  const int tid = threadIdx.x, w = tid >> 5, lane = tid & 31;
  const int hi = lane >> 4, ln = lane & 15;
  const size_t ldq = 3 * D_MODEL;

  // stage q+u, q+v (q = last QLEN rows of qkv's q-third)
  for (int idx = tid; idx < QT * 64; idx += 256) {
    int i = idx >> 6, d = idx & 63;
    float q = (float)qkv[(size_t)((i0 + i + MLEN) * BATCH + b) * ldq + h * 64 + d];
    qu[idx] = (h16)(q + u[h * 64 + d]);
    qv[idx] = (h16)(q + v[h * 64 + d]);
  }
  __syncthreads();

  // ---- score phase: wave w owns row-group mi=w&1, k-tiles jt = (w>>1)+4t ----
  const int mi = w & 1;
  const int ibase = i0 + mi * 16;
  h16*   ktw = kt + w * 16 * 64;
  h16*   rtw = rt + w * 32 * 64;
  float* bdw = bd + w * 16 * 32;
  for (int t = 0; t < 16; ++t) {
    const int jt = (w >> 1) + 4 * t;  // 0..63
    const int j0 = jt * 16;
    {  // K tile 16x64 via async DMA (lane: row lane/2, 32 contiguous halves)
      int r = lane >> 1, c = (lane & 1) * 32;
      const h16* src = &qkv[(size_t)((j0 + r) * BATCH + b) * ldq + D_MODEL + h * 64 + c];
#pragma unroll
      for (int q8 = 0; q8 < 32; q8 += 8)
        lds_async_b128(&ktw[r * 64 + c + q8], &src[q8]);
    }
    const int p0 = j0 - ibase + (QLEN - 16);  // >= 0 always
    {  // R tile 32x64 via async DMA (lane: one row, clamp OOB rows -- masked)
      int p = p0 + lane; if (p > KLEN - 1) p = KLEN - 1;
      const h16* src = &rk[(size_t)p * D_MODEL + h * 64];
#pragma unroll
      for (int c = 0; c < 64; c += 8)
        lds_async_b128(&rtw[lane * 64 + c], &src[c]);
    }
    wait_async();                     // per-wave tiles: ASYNCcnt==0 suffices
    __builtin_amdgcn_wave_barrier();

    v8f ac = {}, bd0 = {}, bd1 = {};
#pragma unroll
    for (int ks = 0; ks < 2; ++ks) {
      v16h aqu, aqv, bk, br0, br1;
#pragma unroll
      for (int e = 0; e < 16; ++e) {
        int ka = 32 * ks + a_kmap(e, hi);
        aqu[e] = qu[(mi * 16 + ln) * 64 + ka];
        aqv[e] = qv[(mi * 16 + ln) * 64 + ka];
        int kb = 32 * ks + hi * 16 + e;  // d index for B fragments
        bk[e]  = ktw[ln * 64 + kb];          // B[d,j] = k[j,d]
        br0[e] = rtw[ln * 64 + kb];          // B[d,c] = r[p0+c,d], c=0..15
        br1[e] = rtw[(16 + ln) * 64 + kb];   // c=16..31
      }
      ac  = wmma_f16(aqu, bk,  ac);
      bd0 = wmma_f16(aqv, br0, bd0);
      bd1 = wmma_f16(aqv, br1, bd1);
    }
    // spill BD block, gather the rel-shift diagonal, mask+scale, store scores
#pragma unroll
    for (int g = 0; g < 8; ++g) {
      int m = hi * 8 + g;
      bdw[m * 32 + ln]      = bd0[g];
      bdw[m * 32 + 16 + ln] = bd1[g];
    }
    __builtin_amdgcn_wave_barrier();
#pragma unroll
    for (int g = 0; g < 8; ++g) {
      int m  = hi * 8 + g;
      int gi = ibase + m, gj = j0 + ln;
      float val = (ac[g] + bdw[m * 32 + (ln - m + 15)]) * 0.125f;
      if (gj > gi + MLEN) val = -1e30f;  // causal mask
      S[(mi * 16 + m) * SLD + gj] = val;
    }
    __builtin_amdgcn_wave_barrier();
  }
  __syncthreads();

  // ---- softmax over 32 rows x 1024 (8 threads per row) ----
  {
    int r = tid >> 3, s = tid & 7;
    float mx = -3.0e38f;
    for (int j = s; j < KLEN; j += 8) mx = fmaxf(mx, S[r * SLD + j]);
    red[r * 8 + s] = mx;
  }
  __syncthreads();
  if (tid < 32) {
    float mx = red[tid * 8];
#pragma unroll
    for (int s = 1; s < 8; ++s) mx = fmaxf(mx, red[tid * 8 + s]);
    rowm[tid] = mx;
  }
  __syncthreads();
  {
    int r = tid >> 3, s = tid & 7;
    float mx = rowm[r], sm = 0.f;
    for (int j = s; j < KLEN; j += 8) {
      float e = __expf(S[r * SLD + j] - mx);
      S[r * SLD + j] = e;
      sm += e;
    }
    red[r * 8 + s] = sm;
  }
  __syncthreads();
  if (tid < 32) {
    float sm = 0.f;
#pragma unroll
    for (int s = 0; s < 8; ++s) sm += red[tid * 8 + s];
    rows[tid] = sm;
  }
  __syncthreads();

  // ---- O = P·V : wave w -> C tile (mi=w&1, ni=w>>1), 1/rowsum on write ----
  const int ni = w >> 1;
  v8f oacc = {};
  for (int k0 = 0; k0 < KLEN; k0 += 32) {
    __syncthreads();
    {  // shared V tile 32x64 via async DMA (one b128 per thread)
      int r = tid >> 3, c = (tid & 7) * 8;
      lds_async_b128(&vt[r * 64 + c],
          &qkv[(size_t)((k0 + r) * BATCH + b) * ldq + 2 * D_MODEL + h * 64 + c]);
    }
    wait_async();
    __syncthreads();
    v16h af, bfv;
#pragma unroll
    for (int e = 0; e < 16; ++e) {
      af[e]  = (h16)S[(mi * 16 + ln) * SLD + k0 + a_kmap(e, hi)];
      bfv[e] = vt[(hi * 16 + e) * 64 + ni * 16 + ln];
    }
    oacc = wmma_f16(af, bfv, oacc);
  }
#pragma unroll
  for (int g = 0; g < 8; ++g) {
    int m = hi * 8 + g;
    int i = i0 + mi * 16 + m;
    float val = oacc[g] / rows[mi * 16 + m];
    out[(size_t)(i * BATCH + b) * D_MODEL + h * 64 + ni * 16 + ln] = (h16)val;
  }
}

// ---------------------------------------------------------------------------
// GRU recurrent step: g(8x3072) = h(8x1024,f16) @ Whh^T (1024x3072,f16).
// One wave per 16-col tile; h rows 8..15 padded with zeros in the fragment.
// W_hh^T (6 MB f16) stays resident in the 192 MB L2 across all 512 steps.
// ---------------------------------------------------------------------------
#define BT_STRIDE 24  // halves; 48B rows keep b128 stores 16B-aligned

__global__ __launch_bounds__(32) void gru_hh_gemm(
    const h16* __restrict__ hh, const h16* __restrict__ Wt,
    float* __restrict__ gout) {
  __shared__ h16 bt[32 * BT_STRIDE];
  const int nt = blockIdx.x;  // 0..191
  const int lane = threadIdx.x;
  const int hi = lane >> 4, ln = lane & 15;
  v8f acc = {};
  for (int k0 = 0; k0 < D_MODEL; k0 += 32) {
    const h16* src = &Wt[(size_t)(k0 + lane) * (3 * D_MODEL) + nt * 16];
    lds_async_b128(&bt[lane * BT_STRIDE + 0], &src[0]);
    lds_async_b128(&bt[lane * BT_STRIDE + 8], &src[8]);
    wait_async();
    __builtin_amdgcn_wave_barrier();
    v16h af, bf;
#pragma unroll
    for (int e = 0; e < 16; ++e) {
      af[e] = (ln < BATCH) ? hh[(size_t)ln * D_MODEL + k0 + a_kmap(e, hi)]
                           : (h16)0.f;
      bf[e] = bt[(hi * 16 + e) * BT_STRIDE + ln];
    }
    acc = wmma_f16(af, bf, acc);
    __builtin_amdgcn_wave_barrier();
  }
#pragma unroll
  for (int g8 = 0; g8 < 8; ++g8) {
    int m = hi * 8 + g8;
    if (m < BATCH) gout[(size_t)m * (3 * D_MODEL) + nt * 16 + ln] = acc[g8];
  }
}

// GRU gates (elementwise, in-place h update) + write output row t
__global__ __launch_bounds__(256) void gru_gate(
    const float* __restrict__ xall,  // 4096 x 3072 (x@Wih^T + b_ih)
    const float* __restrict__ g,     // 8 x 3072
    const float* __restrict__ bhh,   // 3072
    float* __restrict__ h, h16* __restrict__ hh,
    float* __restrict__ out, int t) {
  int idx = blockIdx.x * 256 + threadIdx.x;
  if (idx >= BATCH * D_MODEL) return;
  int bb = idx >> 10, f = idx & (D_MODEL - 1);
  const float* xr = &xall[(size_t)(t * BATCH + bb) * (3 * D_MODEL)];
  const float* gr = &g[(size_t)bb * (3 * D_MODEL)];
  float r = 1.f / (1.f + __expf(-(xr[f] + gr[f] + bhh[f])));
  float z = 1.f / (1.f + __expf(-(xr[D_MODEL + f] + gr[D_MODEL + f] + bhh[D_MODEL + f])));
  float n = tanhf(xr[2 * D_MODEL + f] + r * (gr[2 * D_MODEL + f] + bhh[2 * D_MODEL + f]));
  float hn = (1.f - z) * n + z * h[idx];
  h[idx]  = hn;
  hh[idx] = (h16)hn;
  out[(size_t)(t * BATCH + bb) * D_MODEL + f] = hn;
}

// ---------------------------------------------------------------------------
// Host-side orchestration
// ---------------------------------------------------------------------------
extern "C" void kernel_launch(void* const* d_in, const int* in_sizes, int n_in,
                              void* d_out, int out_size, void* d_ws, size_t ws_size,
                              hipStream_t stream) {
  (void)in_sizes; (void)n_in; (void)out_size; (void)ws_size;
  const float* inputs = (const float*)d_in[0];
  const float* r_in   = (const float*)d_in[1];
  const float* u_in   = (const float*)d_in[2];
  const float* v_in   = (const float*)d_in[3];
  const float* mem_in = (const float*)d_in[4];
  const float* W_qkv  = (const float*)d_in[5];
  const float* W_r    = (const float*)d_in[6];
  const float* W_o    = (const float*)d_in[7];
  const float* W_ih   = (const float*)d_in[8];
  const float* W_hh   = (const float*)d_in[9];
  const float* b_ih   = (const float*)d_in[10];
  const float* b_hh   = (const float*)d_in[11];
  const float* h0     = (const float*)d_in[12];
  float* out = (float*)d_out;

  // ---- carve workspace (256B aligned) ----
  size_t off = 0;
  auto carve = [&](size_t bytes) {
    void* p = (char*)d_ws + off;
    off += (bytes + 255) & ~(size_t)255;
    return p;
  };
  h16*   cat_h  = (h16*)carve((size_t)KLEN * BATCH * D_MODEL * 2);      // 16 MB
  h16*   wqkv_h = (h16*)carve((size_t)D_MODEL * 3 * D_MODEL * 2);       //  6 MB
  h16*   wr_h   = (h16*)carve((size_t)D_MODEL * D_MODEL * 2);           //  2 MB
  h16*   wo_h   = (h16*)carve((size_t)D_MODEL * D_MODEL * 2);           //  2 MB
  h16*   wih_t  = (h16*)carve((size_t)D_MODEL * 3 * D_MODEL * 2);       //  6 MB
  h16*   whh_t  = (h16*)carve((size_t)D_MODEL * 3 * D_MODEL * 2);       //  6 MB
  h16*   r_h    = (h16*)carve((size_t)KLEN * D_MODEL * 2);              //  2 MB
  h16*   qkv_h  = (h16*)carve((size_t)KLEN * BATCH * 3 * D_MODEL * 2);  // 48 MB
  h16*   rk_h   = (h16*)carve((size_t)KLEN * D_MODEL * 2);              //  2 MB
  h16*   attn_h = (h16*)carve((size_t)QLEN * BATCH * D_MODEL * 2);      //  8 MB
  h16*   y_h    = (h16*)carve((size_t)QLEN * BATCH * D_MODEL * 2);      //  8 MB
  float* xall   = (float*)carve((size_t)QLEN * BATCH * 3 * D_MODEL * 4);// 48 MB
  float* gbuf   = (float*)carve((size_t)BATCH * 3 * D_MODEL * 4);
  float* hbuf   = (float*)carve((size_t)BATCH * D_MODEL * 4);
  h16*   h_h    = (h16*)carve((size_t)BATCH * D_MODEL * 2);

  // ---- prep: f16 conversions ----
  {
    int n = KLEN * BATCH * D_MODEL;
    build_cat<<<(n + 255) / 256, 256, 0, stream>>>(mem_in, inputs, cat_h, n);
  }
  {
    int n = D_MODEL * 3 * D_MODEL;
    cvt_f32_f16<<<(n + 255) / 256, 256, 0, stream>>>(W_qkv, wqkv_h, n);
  }
  {
    int n = D_MODEL * D_MODEL;
    cvt_f32_f16<<<(n + 255) / 256, 256, 0, stream>>>(W_r, wr_h, n);
    cvt_f32_f16<<<(n + 255) / 256, 256, 0, stream>>>(W_o, wo_h, n);
  }
  {
    int n = KLEN * D_MODEL;  // r has middle dim 1 -> flat copy
    cvt_f32_f16<<<(n + 255) / 256, 256, 0, stream>>>(r_in, r_h, n);
  }
  {
    int n = 3 * D_MODEL * D_MODEL;
    cvt_transpose_f16<<<(n + 255) / 256, 256, 0, stream>>>(W_ih, wih_t, 3 * D_MODEL, D_MODEL);
    cvt_transpose_f16<<<(n + 255) / 256, 256, 0, stream>>>(W_hh, whh_t, 3 * D_MODEL, D_MODEL);
  }

  // ---- big GEMMs ----
  // qkv = cat @ W_qkv : 8192 x 3072 x 1024
  wmma_gemm<true><<<dim3(3 * D_MODEL / 64, KLEN * BATCH / 64), 256, 0, stream>>>(
      cat_h, wqkv_h, qkv_h, nullptr, KLEN * BATCH, 3 * D_MODEL, D_MODEL);
  // r_head_k = r @ W_r : 1024 x 1024 x 1024
  wmma_gemm<true><<<dim3(D_MODEL / 64, KLEN / 64), 256, 0, stream>>>(
      r_h, wr_h, rk_h, nullptr, KLEN, D_MODEL, D_MODEL);

  // ---- fused rel-attention ----
  (void)hipFuncSetAttribute((const void*)attn_fused,
                            hipFuncAttributeMaxDynamicSharedMemorySize,
                            (int)ATTN_SMEM);
  attn_fused<<<dim3(QLEN / QT, BATCH, N_HEADS), 256, ATTN_SMEM, stream>>>(
      qkv_h, rk_h, u_in, v_in, attn_h);

  // y = attn_vec @ W_o : 4096 x 1024 x 1024
  wmma_gemm<true><<<dim3(D_MODEL / 64, QLEN * BATCH / 64), 256, 0, stream>>>(
      attn_h, wo_h, y_h, nullptr, QLEN * BATCH, D_MODEL, D_MODEL);
  // xall = y @ W_ih^T + b_ih : 4096 x 3072 x 1024 (f32 out)
  wmma_gemm<false><<<dim3(3 * D_MODEL / 64, QLEN * BATCH / 64), 256, 0, stream>>>(
      y_h, wih_t, xall, b_ih, QLEN * BATCH, 3 * D_MODEL, D_MODEL);

  // ---- GRU scan (512 sequential steps; W_hh^T is L2-resident) ----
  {
    int n = BATCH * D_MODEL;
    init_h<<<(n + 255) / 256, 256, 0, stream>>>(h0, hbuf, h_h, n);
  }
  for (int t = 0; t < QLEN; ++t) {
    gru_hh_gemm<<<3 * D_MODEL / 16, 32, 0, stream>>>(h_h, whh_t, gbuf);
    gru_gate<<<(BATCH * D_MODEL + 255) / 256, 256, 0, stream>>>(
        xall, gbuf, b_hh, hbuf, h_h, out, t);
  }
}
